// LocalFeatureTransformer_81595788689435
// MI455X (gfx1250) — compile-verified, matrix operand order checked by reading, not run on hardware
//
#include <hip/hip_runtime.h>

// ---------------- problem constants (fixed by reference setup_inputs) -------
constexpr int nB = 2;          // batch
constexpr int nL = 4800;       // tokens feat0
constexpr int nS = 4800;       // tokens feat1
constexpr int nC = 256;        // d_model
constexpr int NH = 8;          // heads
constexpr int HD = 32;         // head dim
constexpr int TOPK = 16;
constexpr int NLAYER = 8;
constexpr int C2 = nC * nC;            // 65536
constexpr int WELEMS = 10 * C2;        // per-layer bf16 weight elems (Wq,Wk,Wv,Wm,W1,W2)
constexpr int TOK = nB * nL;           // 9600 tokens per side

typedef __attribute__((ext_vector_type(16))) __bf16 v16bf;
typedef __attribute__((ext_vector_type(8)))  float  v8f;

// ---------------------------------------------------------------------------
// Weight convert: src f32 [K x N] row-major  ->  dst bf16 [N x K] (B^T layout)
// ---------------------------------------------------------------------------
__global__ void wconv_kernel(const float* __restrict__ src, __bf16* __restrict__ dst,
                             int K, int Nn) {
  int i = blockIdx.x * 256 + threadIdx.x;
  if (i >= K * Nn) return;
  int n = i / K, k = i % K;
  dst[i] = (__bf16)src[(size_t)k * Nn + n];
}

// Elementwise f32 -> bf16 (same layout); for activation/feature buffers.
__global__ void fcvt_kernel(const float* __restrict__ src, __bf16* __restrict__ dst,
                            int n) {
  int i = blockIdx.x * 256 + threadIdx.x;
  if (i >= n) return;
  dst[i] = (__bf16)src[i];
}

// ---------------------------------------------------------------------------
// WMMA GEMM:  D[M x Nn] = alpha * A[M x Kd] * B^T,   B given as [Nn x Kd].
//   A and B independently bf16 (preconverted) or f32 (convert in-loop).
//   One wave -> 16(M) x 64(N) tile via 4x v_wmma_f32_16x16x32_bf16.
//   Block = 4 waves -> 64x64 tile. grid.z = batch.
// ---------------------------------------------------------------------------
template <bool A_BF16, bool B_BF16, bool RELU>
__global__ void gemm_wmma(const void* __restrict__ Ap, const void* __restrict__ Bp,
                          float* __restrict__ D, int M, int Nn, int Kd,
                          size_t strideA, size_t strideB, size_t strideD, float alpha) {
  int wave = threadIdx.x >> 5;
  int lane = threadIdx.x & 31;
  int r = lane & 15, hi = lane >> 4;
  int m0 = blockIdx.y * 64 + wave * 16;
  int n0 = blockIdx.x * 64;
  int zb = blockIdx.z;
  const __bf16* A16 = (const __bf16*)Ap + (size_t)zb * strideA;
  const float*  A32 = (const float*)Ap + (size_t)zb * strideA;
  const __bf16* B16 = (const __bf16*)Bp + (size_t)zb * strideB;
  const float*  B32 = (const float*)Bp + (size_t)zb * strideB;
  float* Db = D + (size_t)zb * strideD;

  v8f acc[4];
#pragma unroll
  for (int t = 0; t < 4; t++)
#pragma unroll
    for (int j = 0; j < 8; j++) acc[t][j] = 0.f;

  for (int k0 = 0; k0 < Kd; k0 += 32) {
    // A fragment: lane holds row (m0+r), K pattern per CDNA5 16-bit A layout
    v16bf a;
    if (A_BF16) {
      const __bf16* arow = A16 + (size_t)(m0 + r) * Kd + k0;
#pragma unroll
      for (int i = 0; i < 16; i++) {
        int kk = ((i < 8) ? i : i + 8) + hi * 8;
        a[i] = arow[kk];
      }
    } else {
      const float* arow = A32 + (size_t)(m0 + r) * Kd + k0;
#pragma unroll
      for (int i = 0; i < 16; i++) {
        int kk = ((i < 8) ? i : i + 8) + hi * 8;
        a[i] = (__bf16)arow[kk];
      }
    }
#pragma unroll
    for (int t = 0; t < 4; t++) {
      int n = n0 + t * 16 + r;
      v16bf b;
      if (B_BF16) {
        const __bf16* brow = B16 + (size_t)n * Kd + k0;
#pragma unroll
        for (int i = 0; i < 16; i++) {
          int kk = ((i < 8) ? i : i + 8) + hi * 8;
          b[i] = brow[kk];
        }
      } else {
        const float* brow = B32 + (size_t)n * Kd + k0;
#pragma unroll
        for (int i = 0; i < 16; i++) {
          int kk = ((i < 8) ? i : i + 8) + hi * 8;
          b[i] = (__bf16)brow[kk];
        }
      }
      acc[t] = __builtin_amdgcn_wmma_f32_16x16x32_bf16(
          false, a, false, b, (short)0, acc[t], false, false);
    }
  }
#pragma unroll
  for (int t = 0; t < 4; t++) {
#pragma unroll
    for (int j = 0; j < 8; j++) {
      int m = m0 + j + hi * 8;
      int n = n0 + t * 16 + r;
      float v = acc[t][j] * alpha;
      if (RELU) v = v > 0.f ? v : 0.f;
      Db[(size_t)m * Nn + n] = v;
    }
  }
}

// ---------------------------------------------------------------------------
// Online softmax stats along one axis of sim[n][nL][nS].
//   Masked-out entries behave as sim = -1e9 (matches reference -INF).
// ---------------------------------------------------------------------------
__global__ void stats_kernel(const float* __restrict__ sim,
                             float* __restrict__ oMax, float* __restrict__ oSum,
                             const unsigned char* __restrict__ mF,
                             const unsigned char* __restrict__ mS,
                             int nFixed, int nScan, size_t strideFixed, size_t strideScan) {
  int wave = threadIdx.x >> 5, lane = threadIdx.x & 31;
  int f = blockIdx.x * 4 + wave;
  int n = f / nFixed, fl = f % nFixed;
  const float* base = sim + (size_t)n * nL * nS + (size_t)fl * strideFixed;
  bool mf = mF[(size_t)n * nFixed + fl] != 0;
  float mx = -3.4e38f, sm = 0.f;
  for (int c = lane; c < nScan; c += 32) {
    float v = (mf && (mS[(size_t)n * nScan + c] != 0)) ? base[(size_t)c * strideScan] : -1e9f;
    float nm = fmaxf(mx, v);
    sm = sm * __expf(mx - nm) + __expf(v - nm);
    mx = nm;
  }
  for (int o = 16; o; o >>= 1) {
    float om = __shfl_xor(mx, o, 32), os = __shfl_xor(sm, o, 32);
    float nm = fmaxf(mx, om);
    sm = sm * __expf(mx - nm) + os * __expf(om - nm);
    mx = nm;
  }
  if (lane == 0) { oMax[f] = mx; oSum[f] = sm; }
}

// ---------------------------------------------------------------------------
// Top-16 of conf = softmax_row * softmax_col along one axis.
//   conf = exp(2v - aMax - bMax[c]) / (aSum * bSum[c]) ; exponent <= 0, no ovfl.
//   Row staged in LDS, 16 rounds of wave argmax. 2 waves / block.
// ---------------------------------------------------------------------------
__global__ void topk_kernel(const float* __restrict__ sim,
                            const float* __restrict__ aMax, const float* __restrict__ aSum,
                            const float* __restrict__ bMax, const float* __restrict__ bSum,
                            const unsigned char* __restrict__ mF,
                            const unsigned char* __restrict__ mS,
                            int* __restrict__ outIdx,
                            int nFixed, int nScan, size_t strideFixed, size_t strideScan) {
  __shared__ float sbuf[2][4800];
  int wave = threadIdx.x >> 5, lane = threadIdx.x & 31;
  int f = blockIdx.x * 2 + wave;
  int n = f / nFixed, fl = f % nFixed;
  const float* base = sim + (size_t)n * nL * nS + (size_t)fl * strideFixed;
  bool mf = mF[(size_t)n * nFixed + fl] != 0;
  float am = aMax[f];
  float ai = 1.f / aSum[f];
  float* row = sbuf[wave];
  for (int c = lane; c < nScan; c += 32) {
    float v = (mf && (mS[(size_t)n * nScan + c] != 0)) ? base[(size_t)c * strideScan] : -1e9f;
    float conf = __expf(2.f * v - am - bMax[(size_t)n * nScan + c]) *
                 ai * (1.f / bSum[(size_t)n * nScan + c]);
    row[c] = conf;
  }
  __syncthreads();
  for (int kk = 0; kk < TOPK; kk++) {
    float bv = -1.f; int bi = 0x7fffffff;
    for (int c = lane; c < nScan; c += 32) {
      float v = row[c];
      if (v > bv || (v == bv && c < bi)) { bv = v; bi = c; }
    }
    for (int o = 16; o; o >>= 1) {
      float ov = __shfl_xor(bv, o, 32);
      int   oi = __shfl_xor(bi, o, 32);
      if (ov > bv || (ov == bv && oi < bi)) { bv = ov; bi = oi; }
    }
    if (lane == 0) {
      outIdx[(size_t)f * TOPK + kk] = bi;
      row[bi] = -3.4e38f;
    }
    __syncthreads();
  }
}

// ---------------------------------------------------------------------------
// LoFTR linear attention over the K=16 gathered tokens.
//   One wave per (token, head); lane = head-dim index d.
//   Q = elu(q)+1 (masked); K = elu(k)+1; v/=16; KV[d][v'] += K_d * V_v'
//   msg = (Q^T KV) * Z * 16  via LDS transpose.
// ---------------------------------------------------------------------------
__global__ void attn_kernel(const float* __restrict__ Qp, const float* __restrict__ Kp,
                            const float* __restrict__ Vp, const int* __restrict__ idx,
                            const unsigned char* __restrict__ mk,
                            float* __restrict__ msg, int tokPerBatch) {
  __shared__ float part[4][32][33];
  int wave = threadIdx.x >> 5, lane = threadIdx.x & 31;
  int w = blockIdx.x * 4 + wave;
  int t = w >> 3;
  int h = w & 7;
  int srcBase = (t / tokPerBatch) * tokPerBatch;   // nL == nS

  float q = Qp[(size_t)t * nC + h * HD + lane];
  float Qf = q > 0.f ? q + 1.f : __expf(q);
  if (!mk[t]) Qf = 0.f;

  float kv[32];
#pragma unroll
  for (int i = 0; i < 32; i++) kv[i] = 0.f;
  float ksum = 0.f;
  for (int j = 0; j < TOPK; j++) {
    int s = srcBase + idx[(size_t)t * TOPK + j];
    float kr = Kp[(size_t)s * nC + h * HD + lane];
    float kk = kr > 0.f ? kr + 1.f : __expf(kr);
    float vv = Vp[(size_t)s * nC + h * HD + lane] * (1.f / 16.f);
    ksum += kk;
#pragma unroll
    for (int vp = 0; vp < 32; vp++) kv[vp] += kk * __shfl(vv, vp, 32);
  }
  float zd = Qf * ksum;
  for (int o = 16; o; o >>= 1) zd += __shfl_xor(zd, o, 32);
  float z = 1.f / (zd + 1e-6f);
#pragma unroll
  for (int vp = 0; vp < 32; vp++) part[wave][lane][vp] = Qf * kv[vp];
  __syncthreads();
  float sum = 0.f;
#pragma unroll
  for (int d = 0; d < 32; d++) sum += part[wave][d][lane];
  msg[(size_t)t * nC + h * HD + lane] = sum * z * 16.f;
}

// ---------------------------------------------------------------------------
// LayerNorm over 256; optional fused residual add (resid += ln(x)).
// ---------------------------------------------------------------------------
__global__ void ln_kernel(const float* __restrict__ in, const float* __restrict__ g,
                          const float* __restrict__ b, float* __restrict__ out,
                          float* __restrict__ resid, int T) {
  int wave = threadIdx.x >> 5, lane = threadIdx.x & 31;
  int t = blockIdx.x * 4 + wave;
  const float* x = in + (size_t)t * nC;
  float loc[8];
  float s = 0.f;
#pragma unroll
  for (int j = 0; j < 8; j++) { loc[j] = x[lane + j * 32]; s += loc[j]; }
  for (int o = 16; o; o >>= 1) s += __shfl_xor(s, o, 32);
  float mean = s * (1.f / 256.f);
  float v = 0.f;
#pragma unroll
  for (int j = 0; j < 8; j++) { float d = loc[j] - mean; v += d * d; }
  for (int o = 16; o; o >>= 1) v += __shfl_xor(v, o, 32);
  float rstd = rsqrtf(v * (1.f / 256.f) + 1e-5f);
#pragma unroll
  for (int j = 0; j < 8; j++) {
    int c = lane + j * 32;
    float y = (loc[j] - mean) * rstd * g[c] + b[c];
    if (resid) resid[(size_t)t * nC + c] += y;
    else       out[(size_t)t * nC + c] = y;
  }
}

// cat[t] = [ x[t] (256) | m[t] (256) ]
__global__ void pack_cat_kernel(const float* __restrict__ x, const float* __restrict__ m,
                                float* __restrict__ cat, int T) {
  int i = blockIdx.x * 256 + threadIdx.x;
  if (i >= T * nC) return;
  int t = i >> 8, c = i & 255;
  cat[(size_t)t * 512 + c] = x[i];
  cat[(size_t)t * 512 + 256 + c] = m[i];
}

// ---------------------------------------------------------------------------
extern "C" void kernel_launch(void* const* d_in, const int* in_sizes, int n_in,
                              void* d_out, int out_size, void* d_ws, size_t ws_size,
                              hipStream_t stream) {
  (void)in_sizes; (void)n_in; (void)out_size; (void)ws_size;
  const float* f0in = (const float*)d_in[0];
  const float* f1in = (const float*)d_in[1];
  const unsigned char* mask0 = (const unsigned char*)d_in[2];
  const unsigned char* mask1 = (const unsigned char*)d_in[3];
  const float* Wq = (const float*)d_in[4];
  const float* Wk = (const float*)d_in[5];
  const float* Wv = (const float*)d_in[6];
  const float* Wm = (const float*)d_in[7];
  const float* W1 = (const float*)d_in[8];
  const float* W2 = (const float*)d_in[9];
  const float* g1 = (const float*)d_in[10];
  const float* b1 = (const float*)d_in[11];
  const float* g2 = (const float*)d_in[12];
  const float* b2 = (const float*)d_in[13];

  float* out0 = (float*)d_out;
  float* out1 = out0 + (size_t)nB * nL * nC;

  // ---- workspace bump allocator (~340 MB total) ----
  char* base = (char*)d_ws;
  size_t off = 0;
  auto alloc = [&](size_t bytes) -> void* {
    void* p = base + off;
    off += (bytes + 255) & ~(size_t)255;
    return p;
  };
  __bf16* wbf = (__bf16*)alloc(sizeof(__bf16) * (size_t)NLAYER * WELEMS);
  float* sim  = (float*)alloc(sizeof(float) * (size_t)nB * nL * nS);
  __bf16* f0b = (__bf16*)alloc(sizeof(__bf16) * (size_t)TOK * nC);
  __bf16* f1b = (__bf16*)alloc(sizeof(__bf16) * (size_t)TOK * nC);
  float* rM   = (float*)alloc(sizeof(float) * nB * nL);
  float* rS   = (float*)alloc(sizeof(float) * nB * nL);
  float* cM   = (float*)alloc(sizeof(float) * nB * nS);
  float* cS   = (float*)alloc(sizeof(float) * nB * nS);
  int*   idx0 = (int*)alloc(sizeof(int) * (size_t)TOK * TOPK);
  int*   idx1 = (int*)alloc(sizeof(int) * (size_t)TOK * TOPK);
  float* q0   = (float*)alloc(sizeof(float) * (size_t)TOK * nC);
  float* kp0  = (float*)alloc(sizeof(float) * (size_t)TOK * nC);
  float* vp0  = (float*)alloc(sizeof(float) * (size_t)TOK * nC);
  float* q1   = (float*)alloc(sizeof(float) * (size_t)TOK * nC);
  float* kp1  = (float*)alloc(sizeof(float) * (size_t)TOK * nC);
  float* vp1  = (float*)alloc(sizeof(float) * (size_t)TOK * nC);
  float* msgb = (float*)alloc(sizeof(float) * (size_t)TOK * nC);
  float* mout = (float*)alloc(sizeof(float) * (size_t)TOK * nC);
  float* cat  = (float*)alloc(sizeof(float) * (size_t)TOK * 2 * nC);
  float* h1   = (float*)alloc(sizeof(float) * (size_t)TOK * 2 * nC);
  float* ffo  = (float*)alloc(sizeof(float) * (size_t)TOK * nC);

  // ---- features -> d_out, updated in place ----
  hipMemcpyAsync(out0, f0in, sizeof(float) * (size_t)nB * nL * nC,
                 hipMemcpyDeviceToDevice, stream);
  hipMemcpyAsync(out1, f1in, sizeof(float) * (size_t)nB * nS * nC,
                 hipMemcpyDeviceToDevice, stream);

  // ---- convert all layer weights to bf16 B^T once ----
  auto cvtW = [&](const float* src, __bf16* dst, int K, int Nn) {
    int tot = K * Nn;
    wconv_kernel<<<(tot + 255) / 256, 256, 0, stream>>>(src, dst, K, Nn);
  };
  for (int i = 0; i < NLAYER; i++) {
    __bf16* wl = wbf + (size_t)i * WELEMS;
    cvtW(Wq + (size_t)i * C2, wl + 0 * C2, nC, nC);
    cvtW(Wk + (size_t)i * C2, wl + 1 * C2, nC, nC);
    cvtW(Wv + (size_t)i * C2, wl + 2 * C2, nC, nC);
    cvtW(Wm + (size_t)i * C2, wl + 3 * C2, nC, nC);
    cvtW(W1 + (size_t)i * 4 * C2, wl + 4 * C2, 2 * nC, 2 * nC);  // 512x512
    cvtW(W2 + (size_t)i * 2 * C2, wl + 8 * C2, 2 * nC, nC);      // 512x256
  }

  auto cvtFeats = [&]() {
    fcvt_kernel<<<(TOK * nC) / 256, 256, 0, stream>>>(out0, f0b, TOK * nC);
    fcvt_kernel<<<(TOK * nC) / 256, 256, 0, stream>>>(out1, f1b, TOK * nC);
  };
  cvtFeats();

  // ---- launch helpers ----
  auto gemmW = [&](const float* A, const __bf16* Bt, float* D, int M, int Nn, int Kd,
                   bool relu) {
    dim3 g(Nn / 64, M / 64, 1), blk(128);
    if (relu)
      gemm_wmma<false, true, true><<<g, blk, 0, stream>>>(A, Bt, D, M, Nn, Kd, 0, 0, 0, 1.f);
    else
      gemm_wmma<false, true, false><<<g, blk, 0, stream>>>(A, Bt, D, M, Nn, Kd, 0, 0, 0, 1.f);
  };
  auto gemmProj = [&](const __bf16* Ab16, const __bf16* Bt, float* D) {
    dim3 g(nC / 64, TOK / 64, 1), blk(128);
    gemm_wmma<true, true, false><<<g, blk, 0, stream>>>(Ab16, Bt, D, TOK, nC, nC,
                                                        0, 0, 0, 1.f);
  };
  auto gemmSim = [&](const __bf16* Ab16, const __bf16* Bb16) {
    dim3 g(nS / 64, nL / 64, nB), blk(128);
    gemm_wmma<true, true, false><<<g, blk, 0, stream>>>(
        Ab16, Bb16, sim, nL, nS, nC, (size_t)nL * nC, (size_t)nS * nC,
        (size_t)nL * nS, 1.f / (nC * 0.1f));
  };
  auto statsRow = [&](const unsigned char* m0p, const unsigned char* m1p) {
    stats_kernel<<<(nB * nL) / 4, 128, 0, stream>>>(sim, rM, rS, m0p, m1p, nL, nS,
                                                    (size_t)nS, (size_t)1);
  };
  auto statsCol = [&](const unsigned char* m0p, const unsigned char* m1p) {
    stats_kernel<<<(nB * nS) / 4, 128, 0, stream>>>(sim, cM, cS, m1p, m0p, nS, nL,
                                                    (size_t)1, (size_t)nS);
  };
  auto topkRow = [&](int* oi, const unsigned char* m0p, const unsigned char* m1p) {
    topk_kernel<<<(nB * nL) / 2, 64, 0, stream>>>(sim, rM, rS, cM, cS, m0p, m1p, oi,
                                                  nL, nS, (size_t)nS, (size_t)1);
  };
  auto topkCol = [&](int* oi, const unsigned char* m0p, const unsigned char* m1p) {
    topk_kernel<<<(nB * nS) / 2, 64, 0, stream>>>(sim, cM, cS, rM, rS, m1p, m0p, oi,
                                                  nS, nL, (size_t)1, (size_t)nS);
  };
  auto doSide = [&](float* x, const float* q, const float* kp, const float* vp,
                    const int* idx, const unsigned char* mk, const __bf16* wl, int layer) {
    attn_kernel<<<(TOK * NH) / 4, 128, 0, stream>>>(q, kp, vp, idx, mk, msgb, nL);
    gemmW(msgb, wl + 3 * C2, mout, TOK, nC, nC, false);            // msg @ Wm
    ln_kernel<<<TOK / 4, 128, 0, stream>>>(mout, g1 + (size_t)layer * nC,
                                           b1 + (size_t)layer * nC, msgb, nullptr, TOK);
    pack_cat_kernel<<<(TOK * nC) / 256, 256, 0, stream>>>(x, msgb, cat, TOK);
    gemmW(cat, wl + 4 * C2, h1, TOK, 2 * nC, 2 * nC, true);        // relu(cat @ W1)
    gemmW(h1, wl + 8 * C2, ffo, TOK, nC, 2 * nC, false);           // @ W2
    ln_kernel<<<TOK / 4, 128, 0, stream>>>(ffo, g2 + (size_t)layer * nC,
                                           b2 + (size_t)layer * nC, nullptr, x, TOK);
  };

  // ---- 8 layers: self, cross, self, cross, ... ----
  for (int i = 0; i < NLAYER; i++) {
    const __bf16* wl = wbf + (size_t)i * WELEMS;
    bool selfL = (i % 2) == 0;
    if (selfL) {
      gemmSim(f0b, f0b);
      statsRow(mask0, mask0); statsCol(mask0, mask0);
      topkRow(idx0, mask0, mask0);
      gemmSim(f1b, f1b);
      statsRow(mask1, mask1); statsCol(mask1, mask1);
      topkRow(idx1, mask1, mask1);
    } else {
      gemmSim(f0b, f1b);
      statsRow(mask0, mask1); statsCol(mask0, mask1);
      topkRow(idx0, mask0, mask1);     // feat0 rows pick feat1 tokens
      topkCol(idx1, mask0, mask1);     // feat1 rows (conf^T) pick feat0 tokens
    }
    // projections from pre-update features (gather commutes with linear proj)
    gemmProj(f0b, wl + 0 * C2, q0);
    gemmProj(f0b, wl + 1 * C2, kp0);
    gemmProj(f0b, wl + 2 * C2, vp0);
    gemmProj(f1b, wl + 0 * C2, q1);
    gemmProj(f1b, wl + 1 * C2, kp1);
    gemmProj(f1b, wl + 2 * C2, vp1);

    doSide(out0, q0, selfL ? kp0 : kp1, selfL ? vp0 : vp1, idx0, mask0, wl, i);
    doSide(out1, q1, selfL ? kp1 : kp0, selfL ? vp1 : vp0, idx1, mask1, wl, i);

    // refresh bf16 feature copies for next layer's GEMMs
    if (i + 1 < NLAYER) cvtFeats();
  }
}
